// DARNN_34634616275452
// MI455X (gfx1250) — matrix operand order repaired and checked by reading, not compiled, not run on hardware
//
#include <hip/hip_runtime.h>
#include <hip/hip_bf16.h>
#include <math.h>

typedef __attribute__((ext_vector_type(16))) _Float16 v16h;
typedef __attribute__((ext_vector_type(8)))  float    v8f;

#define BT   16      // batch rows per block
#define TT   64      // timesteps (TM1)
#define NIN  63      // input features
#define NKP  64      // NIN padded to 64
#define HH   128     // hidden (H == D == E)
#define G4   512     // 4*H gate width
#define NTHR 256     // 8 wave32s

// ---- LDS layout (bytes), total 282816 < 320KB/WGP ----
#define OFF_WBIG 0        // 512*128 f16   (enc_Whh / dec_Whh)        131072
#define OFF_WSM  131072   // 512*64  f16   (enc_Wih pad / W1h+W1c / W1e) 65536
#define OFF_G    196608   // 16*512  f32   gate GEMM result            32768
#define OFF_HF   229376   // 16*128  f32   h                            8192
#define OFF_CF   237568   // 16*128  f32   c                            8192
#define OFF_H16  245760   // 16*128  f16   h (WMMA A)                   4096
#define OFF_C16  249856   // 16*128  f16   c (WMMA A, decoder)          4096
#define OFF_WI   253952   // 16*64   f16   weighted input               2048
#define OFF_SS   256000   // 16*64   f32   series_score / dec scores    4096
#define OFF_AT   260096   // 16*64   f32   attn workspace               4096
#define OFF_RED  264192   // 16*16   f32   reductions                   1024
#define OFF_SC0  265216   // 16 f32
#define OFF_SC1  265280   // 16 f32
#define OFF_SC2  265344   // 16 f32
#define OFF_HCP  265408   // 16*128  f32   h@W1h + c@W1c                8192
#define OFF_CTX  273600   // 16*128  f32   context                      8192
#define OFF_WV   281792   // 256 f32      Wh|Wc                         1024
#define SMEM_BYTES 282816

__device__ __forceinline__ float sigf(float x) { return 1.0f / (1.0f + expf(-x)); }

// A fragment (16x32 f16, row-major [m][K]): lane m=l&15, chunks at kb and kb+16,
// kb = (l>>4)*8 + 32*kstep.  Two b128 loads.
__device__ __forceinline__ v16h load_frag_a(const _Float16* p) {
    union { v16h v; uint4 q[2]; } r;
    r.q[0] = *(const uint4*)(p);
    r.q[1] = *(const uint4*)(p + 16);
    return r.v;
}
// B fragment (32x16 f16, weights stored row-major [n][K], B[k][n]=W[n][k]):
// lane n=l&15, chunks at kb and kb+16, kb = (l>>4)*8 + 32*kstep. Two b128 loads.
__device__ __forceinline__ v16h load_frag_b(const _Float16* p) {
    union { v16h v; uint4 q[2]; } r;
    r.q[0] = *(const uint4*)(p);
    r.q[1] = *(const uint4*)(p + 16);
    return r.v;
}
__device__ __forceinline__ v8f wmma_f16(v16h a, v16h b, v8f c) {
    return __builtin_amdgcn_wmma_f32_16x16x32_f16(false, a, false, b, (short)0, c, false, false);
}

__global__ __launch_bounds__(NTHR)
void darnn_kernel(const float* __restrict__ x,          // [512][64][64]
                  const float* __restrict__ enc_attn_W, // [320] = Wh|Wc|Wt
                  const float* __restrict__ enc_attn_b, // [1]
                  const float* __restrict__ enc_Wih,    // [512][63]
                  const float* __restrict__ enc_Whh,    // [512][128]
                  const float* __restrict__ enc_bih,    // [512]
                  const float* __restrict__ enc_bhh,    // [512]
                  const float* __restrict__ dec_W1,     // [128][384] = W1h|W1c|W1e
                  const float* __restrict__ dec_b1,     // [128]
                  const float* __restrict__ dec_W2,     // [128]
                  const float* __restrict__ dec_b2,     // [1]
                  const float* __restrict__ dec_Wih,    // [512]
                  const float* __restrict__ dec_Whh,    // [512][128]
                  const float* __restrict__ dec_bih,    // [512]
                  const float* __restrict__ dec_bhh,    // [512]
                  const float* __restrict__ fc_W,       // [129]
                  const float* __restrict__ fc_b,       // [1]
                  const float* __restrict__ fcf_W,      // [256]
                  const float* __restrict__ fcf_b,      // [1]
                  float* __restrict__ out,              // [512]
                  float* __restrict__ ws_encf,          // [512*64*128] f32
                  float* __restrict__ ws_proj,          // [512*64*128] f32
                  _Float16* __restrict__ ws_enc16)      // [512*64*128] f16
{
    extern __shared__ char smem[];
    _Float16* Wbig  = (_Float16*)(smem + OFF_WBIG);
    _Float16* Wsm   = (_Float16*)(smem + OFF_WSM);
    float*    gld   = (float*)(smem + OFF_G);
    float*    hf    = (float*)(smem + OFF_HF);
    float*    cfv   = (float*)(smem + OFF_CF);
    _Float16* h16   = (_Float16*)(smem + OFF_H16);
    _Float16* c16   = (_Float16*)(smem + OFF_C16);
    _Float16* wi16  = (_Float16*)(smem + OFF_WI);
    float*    sscr  = (float*)(smem + OFF_SS);
    float*    attnv = (float*)(smem + OFF_AT);
    float*    red   = (float*)(smem + OFF_RED);
    float*    scal0 = (float*)(smem + OFF_SC0);
    float*    scal1 = (float*)(smem + OFF_SC1);
    float*    scal2 = (float*)(smem + OFF_SC2);
    float*    hcp   = (float*)(smem + OFF_HCP);
    float*    ctx   = (float*)(smem + OFF_CTX);
    float*    WhWc  = (float*)(smem + OFF_WV);

    const int tid  = threadIdx.x;
    const int lane = tid & 31;
    // Force wave index into an SGPR: scalar loop control around WMMA keeps
    // EXEC all-1s (ISA requirement for WMMA) and removes per-lane trip counts.
    const int wave = __builtin_amdgcn_readfirstlane(tid >> 5);
    const int blk  = blockIdx.x;
    const int b0   = blk * BT;

    const float* xblk = x + (size_t)b0 * TT * 64;
    float*    encf  = ws_encf  + (size_t)b0 * TT * HH;
    float*    projf = ws_proj  + (size_t)b0 * TT * HH;
    _Float16* enc16 = ws_enc16 + (size_t)b0 * TT * HH;

    const int m8  = lane & 15;               // A-row / B-col within tile
    const int kbA = ((lane >> 4) << 3);      // A fragment K base
    const int kbB = ((lane >> 4) << 3);      // B fragment K base
    const int mb  = ((lane >> 4) << 3);      // C/D M base

    // ---------------- Phase A: encoder weights + init ----------------
    for (int i = tid; i < 2 * HH; i += NTHR) WhWc[i] = enc_attn_W[i];
    for (int i = tid; i < G4 * HH; i += NTHR) Wbig[i] = (_Float16)enc_Whh[i];
    for (int i = tid; i < G4 * NKP; i += NTHR) {
        int n = i >> 6, k = i & 63;
        Wsm[i] = (k < NIN) ? (_Float16)enc_Wih[n * NIN + k] : (_Float16)0.f;
    }
    for (int i = tid; i < BT * HH; i += NTHR) {
        hf[i] = 0.f; cfv[i] = 0.f; h16[i] = (_Float16)0.f; c16[i] = (_Float16)0.f;
    }
    const float attn_b = enc_attn_b[0];
    for (int p = tid; p < BT * 64; p += NTHR) {        // series_score
        int b = p >> 6, n = p & 63;
        float s = 0.f;
        if (n < NIN) {
            for (int t = 0; t < TT; ++t)
                s += xblk[(b * TT + t) * 64 + (n + 1)] * enc_attn_W[2 * HH + t];
            s += attn_b;
        }
        sscr[p] = s;
    }
    __syncthreads();

    // ---------------- Encoder: 64 recurrent steps ----------------
    const int rb  = tid >> 4;   // row for 16-thread-per-row work
    const int grp = tid & 15;
    for (int t = 0; t < TT; ++t) {
        // score_b = h.Wh + c.Wc
        {
            float part = 0.f;
            for (int k = grp * 8; k < grp * 8 + 8; ++k)
                part += hf[rb * HH + k] * WhWc[k] + cfv[rb * HH + k] * WhWc[HH + k];
            red[rb * 16 + grp] = part;
        }
        __syncthreads();
        if (grp == 0) { float s = 0.f; for (int i = 0; i < 16; ++i) s += red[rb * 16 + i]; scal0[rb] = s; }
        __syncthreads();
        // softmax over features (63), then wi = attn * x_t
        float mx = -3.0e38f;
        for (int j = 0; j < 4; ++j) {
            int n = grp + 16 * j;
            if (n < NIN) {
                float v = scal0[rb] + sscr[rb * 64 + n];
                attnv[rb * 64 + n] = v;
                mx = fmaxf(mx, v);
            }
        }
        red[rb * 16 + grp] = mx;
        __syncthreads();
        if (grp == 0) { float mm = -3.0e38f; for (int i = 0; i < 16; ++i) mm = fmaxf(mm, red[rb * 16 + i]); scal1[rb] = mm; }
        __syncthreads();
        float ssum = 0.f;
        for (int j = 0; j < 4; ++j) {
            int n = grp + 16 * j;
            if (n < NIN) { float e = expf(attnv[rb * 64 + n] - scal1[rb]); attnv[rb * 64 + n] = e; ssum += e; }
        }
        red[rb * 16 + grp] = ssum;
        __syncthreads();
        if (grp == 0) { float s = 0.f; for (int i = 0; i < 16; ++i) s += red[rb * 16 + i]; scal2[rb] = s; }
        __syncthreads();
        {
            float inv = 1.f / scal2[rb];
            for (int j = 0; j < 4; ++j) {
                int n = grp + 16 * j;
                float w = (n < NIN) ? attnv[rb * 64 + n] * inv * xblk[(rb * TT + t) * 64 + n + 1] : 0.f;
                wi16[rb * 64 + n] = (_Float16)w;
            }
        }
        __syncthreads();

        // gates = wi @ Wih^T + h @ Whh^T     (16 x 512, WMMA f16->f32)
        {
            v16h ah[4], aw[2];
#pragma unroll
            for (int ks = 0; ks < 4; ++ks) ah[ks] = load_frag_a(h16 + m8 * HH + kbA + 32 * ks);
#pragma unroll
            for (int ks = 0; ks < 2; ++ks) aw[ks] = load_frag_a(wi16 + m8 * NKP + kbA + 32 * ks);
#pragma unroll
            for (int ni = 0; ni < 4; ++ni) {
                const int nt = wave + 8 * ni;         // uniform (SGPR) tile index
                const int nrow = nt * 16 + m8;
                v8f acc = {0.f, 0.f, 0.f, 0.f, 0.f, 0.f, 0.f, 0.f};
#pragma unroll
                for (int ks = 0; ks < 4; ++ks)
                    acc = wmma_f16(ah[ks], load_frag_b(Wbig + nrow * HH + kbB + 32 * ks), acc);
#pragma unroll
                for (int ks = 0; ks < 2; ++ks)
                    acc = wmma_f16(aw[ks], load_frag_b(Wsm + nrow * NKP + kbB + 32 * ks), acc);
#pragma unroll
                for (int r = 0; r < 8; ++r)
                    gld[(mb + r) * G4 + nrow] = acc[r];   // [m][n]
            }
        }
        __syncthreads();
        // gate nonlinearity, state update, emit encoded h (f32 + f16)
        for (int p = tid; p < BT * HH; p += NTHR) {
            int bb = p >> 7, j = p & 127;
            float gi = gld[bb * G4 + j]       + enc_bih[j]       + enc_bhh[j];
            float gf = gld[bb * G4 + 128 + j] + enc_bih[128 + j] + enc_bhh[128 + j];
            float gg = gld[bb * G4 + 256 + j] + enc_bih[256 + j] + enc_bhh[256 + j];
            float go = gld[bb * G4 + 384 + j] + enc_bih[384 + j] + enc_bhh[384 + j];
            float cn = sigf(gf) * cfv[p] + sigf(gi) * tanhf(gg);
            float hn = sigf(go) * tanhf(cn);
            cfv[p] = cn; hf[p] = hn; h16[p] = (_Float16)hn;
            encf[(bb * TT + t) * HH + j]  = hn;
            enc16[(bb * TT + t) * HH + j] = (_Float16)hn;
        }
        __syncthreads();
    }

    // ---------------- Phase B: enc_proj = enc @ W1e^T + b1  ([1024,128]x[128,128]) ----------------
    for (int i = tid; i < HH * HH; i += NTHR) {
        int f = i >> 7, k = i & 127;
        Wsm[i] = (_Float16)dec_W1[f * 384 + 256 + k];   // W1e row-major [f][k]
    }
    __threadfence_block();
    __syncthreads();
    for (int it = 0; it < 64; ++it) {                  // 512 tiles / 8 waves
        const int idx = wave + 8 * it;                 // uniform
        const int mt = idx >> 3, nt = idx & 7;
        const int arow = mt * 16 + m8;
        const int nrow = nt * 16 + m8;
        v8f acc = {0.f, 0.f, 0.f, 0.f, 0.f, 0.f, 0.f, 0.f};
#pragma unroll
        for (int ks = 0; ks < 4; ++ks) {
            v16h a  = load_frag_a(enc16 + arow * HH + kbA + 32 * ks);
            v16h bf = load_frag_b(Wsm + nrow * HH + kbB + 32 * ks);
            acc = wmma_f16(a, bf, acc);
        }
        const float bn = dec_b1[nrow];
#pragma unroll
        for (int r = 0; r < 8; ++r)
            projf[(mt * 16 + mb + r) * HH + nrow] = acc[r] + bn;
    }
    __syncthreads();

    // ---------------- Phase C: decoder weights + reset state ----------------
    for (int i = tid; i < G4 * HH; i += NTHR) Wbig[i] = (_Float16)dec_Whh[i];
    for (int i = tid; i < HH * HH; i += NTHR) {
        int f = i >> 7, k = i & 127;
        Wsm[i]           = (_Float16)dec_W1[f * 384 + k];        // W1h
        Wsm[HH * HH + i] = (_Float16)dec_W1[f * 384 + 128 + k];  // W1c
    }
    for (int i = tid; i < BT * HH; i += NTHR) {
        hf[i] = 0.f; cfv[i] = 0.f; ctx[i] = 0.f;
        h16[i] = (_Float16)0.f; c16[i] = (_Float16)0.f;
    }
    __threadfence_block();
    __syncthreads();

    // ---------------- Decoder: 64 recurrent steps ----------------
    for (int t = 0; t < TT; ++t) {
        // hcp = h @ W1h^T + c @ W1c^T   (16 x 128)
        {
            const int nrow = wave * 16 + m8;
            v8f acc = {0.f, 0.f, 0.f, 0.f, 0.f, 0.f, 0.f, 0.f};
#pragma unroll
            for (int ks = 0; ks < 4; ++ks) {
                acc = wmma_f16(load_frag_a(h16 + m8 * HH + kbA + 32 * ks),
                               load_frag_b(Wsm + nrow * HH + kbB + 32 * ks), acc);
                acc = wmma_f16(load_frag_a(c16 + m8 * HH + kbA + 32 * ks),
                               load_frag_b(Wsm + HH * HH + nrow * HH + kbB + 32 * ks), acc);
            }
#pragma unroll
            for (int r = 0; r < 8; ++r) hcp[(mb + r) * HH + nrow] = acc[r];
        }
        __syncthreads();
        // scores over time: sum_e tanh(proj + hcp) * W2
        const float b2v = dec_b2[0];
        for (int p = tid; p < BT * TT; p += NTHR) {
            int b = p >> 6, tt = p & 63;
            const float* pr = projf + (b * TT + tt) * HH;
            const float* hp = hcp + b * HH;
            float s = 0.f;
            for (int e = 0; e < HH; ++e) s += tanhf(pr[e] + hp[e]) * dec_W2[e];
            sscr[p] = s + b2v;
        }
        __syncthreads();
        // softmax over 64 timesteps
        float mx = -3.0e38f;
        for (int j = 0; j < 4; ++j) mx = fmaxf(mx, sscr[rb * 64 + grp + 16 * j]);
        red[rb * 16 + grp] = mx;
        __syncthreads();
        if (grp == 0) { float mm = -3.0e38f; for (int i = 0; i < 16; ++i) mm = fmaxf(mm, red[rb * 16 + i]); scal1[rb] = mm; }
        __syncthreads();
        float ssum = 0.f;
        for (int j = 0; j < 4; ++j) {
            int n = grp + 16 * j;
            float e = expf(sscr[rb * 64 + n] - scal1[rb]);
            attnv[rb * 64 + n] = e; ssum += e;
        }
        red[rb * 16 + grp] = ssum;
        __syncthreads();
        if (grp == 0) { float s = 0.f; for (int i = 0; i < 16; ++i) s += red[rb * 16 + i]; scal2[rb] = s; }
        __syncthreads();
        {
            float inv = 1.f / scal2[rb];
            for (int j = 0; j < 4; ++j) attnv[rb * 64 + grp + 16 * j] *= inv;
        }
        __syncthreads();
        // context = attn @ input_encoded   (reads L2-resident encf)
        for (int p = tid; p < BT * HH; p += NTHR) {
            int bb = p >> 7, e = p & 127;
            const float* ef = encf + bb * TT * HH + e;
            float acc = 0.f;
            for (int tt = 0; tt < TT; ++tt) acc += attnv[bb * 64 + tt] * ef[tt * HH];
            ctx[p] = acc;
        }
        __syncthreads();
        // y_tilde = [context, y_t] @ fc_W^T + fc_b
        {
            float part = 0.f;
            for (int k = grp * 8; k < grp * 8 + 8; ++k) part += fc_W[k] * ctx[rb * HH + k];
            red[rb * 16 + grp] = part;
        }
        __syncthreads();
        if (grp == 0) {
            float s = 0.f; for (int i = 0; i < 16; ++i) s += red[rb * 16 + i];
            scal0[rb] = s + fc_W[HH] * xblk[(rb * TT + t) * 64 + 0] + fc_b[0];
        }
        __syncthreads();
        // gates = h @ dec_Whh^T  (16 x 512 WMMA); y_tilde*Wih + biases folded below
        {
            v16h ah[4];
#pragma unroll
            for (int ks = 0; ks < 4; ++ks) ah[ks] = load_frag_a(h16 + m8 * HH + kbA + 32 * ks);
#pragma unroll
            for (int ni = 0; ni < 4; ++ni) {
                const int nt = wave + 8 * ni;         // uniform (SGPR) tile index
                const int nrow = nt * 16 + m8;
                v8f acc = {0.f, 0.f, 0.f, 0.f, 0.f, 0.f, 0.f, 0.f};
#pragma unroll
                for (int ks = 0; ks < 4; ++ks)
                    acc = wmma_f16(ah[ks], load_frag_b(Wbig + nrow * HH + kbB + 32 * ks), acc);
#pragma unroll
                for (int r = 0; r < 8; ++r) gld[(mb + r) * G4 + nrow] = acc[r];
            }
        }
        __syncthreads();
        for (int p = tid; p < BT * HH; p += NTHR) {
            int bb = p >> 7, j = p & 127;
            float yt = scal0[bb];
            float gi = gld[bb * G4 + j]       + yt * dec_Wih[j]       + dec_bih[j]       + dec_bhh[j];
            float gf = gld[bb * G4 + 128 + j] + yt * dec_Wih[128 + j] + dec_bih[128 + j] + dec_bhh[128 + j];
            float gg = gld[bb * G4 + 256 + j] + yt * dec_Wih[256 + j] + dec_bih[256 + j] + dec_bhh[256 + j];
            float go = gld[bb * G4 + 384 + j] + yt * dec_Wih[384 + j] + dec_bih[384 + j] + dec_bhh[384 + j];
            float cn = sigf(gf) * cfv[p] + sigf(gi) * tanhf(gg);
            float hn = sigf(go) * tanhf(cn);
            cfv[p] = cn; hf[p] = hn;
            h16[p] = (_Float16)hn; c16[p] = (_Float16)cn;
        }
        __syncthreads();
    }

    // ---------------- Final: out = [h, context] @ fcf_W^T + fcf_b ----------------
    {
        float part = 0.f;
        for (int k = grp * 8; k < grp * 8 + 8; ++k)
            part += fcf_W[k] * hf[rb * HH + k] + fcf_W[HH + k] * ctx[rb * HH + k];
        red[rb * 16 + grp] = part;
    }
    __syncthreads();
    if (grp == 0) {
        float s = 0.f; for (int i = 0; i < 16; ++i) s += red[rb * 16 + i];
        out[b0 + rb] = s + fcf_b[0];
    }
}

extern "C" void kernel_launch(void* const* d_in, const int* in_sizes, int n_in,
                              void* d_out, int out_size, void* d_ws, size_t ws_size,
                              hipStream_t stream) {
    (void)in_sizes; (void)n_in; (void)out_size; (void)ws_size;
    const float* x          = (const float*)d_in[0];
    const float* enc_attn_W = (const float*)d_in[1];
    const float* enc_attn_b = (const float*)d_in[2];
    const float* enc_Wih    = (const float*)d_in[3];
    const float* enc_Whh    = (const float*)d_in[4];
    const float* enc_bih    = (const float*)d_in[5];
    const float* enc_bhh    = (const float*)d_in[6];
    const float* dec_W1     = (const float*)d_in[7];
    const float* dec_b1     = (const float*)d_in[8];
    const float* dec_W2     = (const float*)d_in[9];
    const float* dec_b2     = (const float*)d_in[10];
    const float* dec_Wih    = (const float*)d_in[11];
    const float* dec_Whh    = (const float*)d_in[12];
    const float* dec_bih    = (const float*)d_in[13];
    const float* dec_bhh    = (const float*)d_in[14];
    const float* fc_W       = (const float*)d_in[15];
    const float* fc_b       = (const float*)d_in[16];
    const float* fcf_W      = (const float*)d_in[17];
    const float* fcf_b      = (const float*)d_in[18];
    float* out = (float*)d_out;

    // workspace: input_encoded f32 (16MB) | enc_proj f32 (16MB) | input_encoded f16 (8MB)
    float* ws = (float*)d_ws;
    const size_t NELT = (size_t)512 * 64 * 128;
    float*    ws_encf  = ws;
    float*    ws_proj  = ws + NELT;
    _Float16* ws_enc16 = (_Float16*)(ws + 2 * NELT);

    (void)hipFuncSetAttribute(reinterpret_cast<const void*>(darnn_kernel),
                              hipFuncAttributeMaxDynamicSharedMemorySize, SMEM_BYTES);
    darnn_kernel<<<dim3(512 / BT), dim3(NTHR), SMEM_BYTES, stream>>>(
        x, enc_attn_W, enc_attn_b, enc_Wih, enc_Whh, enc_bih, enc_bhh,
        dec_W1, dec_b1, dec_W2, dec_b2, dec_Wih, dec_Whh, dec_bih, dec_bhh,
        fc_W, fc_b, fcf_W, fcf_b, out, ws_encf, ws_proj, ws_enc16);
}
